// STModel_57604101374610
// MI455X (gfx1250) — compile-verified, matrix operand order checked by reading, not run on hardware
//
#include <hip/hip_runtime.h>
#include <hip/hip_bf16.h>

typedef __attribute__((ext_vector_type(16))) _Float16 v16h;
typedef __attribute__((ext_vector_type(8)))  _Float16 v8h;
typedef __attribute__((ext_vector_type(8)))  float    v8f;

#define NN 50000
#define EE 600000
#define TT 24

// Branchless fast transcendentals (v_exp_f32 + v_rcp_f32, no libm branches).
__device__ __forceinline__ float sigm(float x) {
    return __builtin_amdgcn_rcpf(1.f + __expf(-x));
}
__device__ __forceinline__ float fast_tanh(float x) {
    float e = __expf(-2.f * fabsf(x));
    float t = (1.f - e) * __builtin_amdgcn_rcpf(1.f + e);
    return copysignf(t, x);
}

// ---------------------------------------------------------------------------
// Weight packing: row-major f32 W[K x 128] -> col-major f16 Bcm[n*Kpad + k]
// ---------------------------------------------------------------------------
__global__ void pack_wkn_kernel(const float* __restrict__ W, _Float16* __restrict__ Bcm,
                                int K, int Kpad) {
    int idx = blockIdx.x * 256 + threadIdx.x;
    if (idx >= Kpad * 128) return;
    int nc = idx / Kpad;
    int k  = idx % Kpad;
    Bcm[(size_t)nc * Kpad + k] = (k < K) ? (_Float16)W[(size_t)k * 128 + nc] : (_Float16)0.f;
}

// alphaW = alpha(KxK) @ featW(Kx128), packed col-major f16 (Kpad == K, mult of 32)
__global__ void alphaW_pack_kernel(const float* __restrict__ alpha, const float* __restrict__ featW,
                                   _Float16* __restrict__ Bcm, int K, int Kpad) {
    int idx = blockIdx.x * 256 + threadIdx.x;
    if (idx >= Kpad * 128) return;
    int nc = idx / Kpad;
    int k  = idx % Kpad;
    float s = 0.f;
    if (k < K) {
        for (int m = 0; m < K; ++m) s += alpha[(size_t)k * K + m] * featW[(size_t)m * 128 + nc];
    }
    Bcm[(size_t)nc * Kpad + k] = (_Float16)s;
}

// ---------------------------------------------------------------------------
// Generic WMMA GEMM: C[n x 128] = relu?(A[n x K] * B + bias[col])
// K, KPAD compile-time: all masking constant-folds, chunk loop fully unrolls.
// block = 256 threads = 8 waves; wave w -> 16-col tile; block -> 16-row stripe
// ---------------------------------------------------------------------------
template <int K, int KPAD, bool RELU>
__global__ __launch_bounds__(256) void gemm_kernel(
    const float* __restrict__ A, int lda,
    const _Float16* __restrict__ Bcm, const float* __restrict__ bias,
    float* __restrict__ C, int ldc, int n) {
    int lane = threadIdx.x & 31;
    int w    = threadIdx.x >> 5;
    int row0 = blockIdx.x * 16;
    int r    = lane & 15;
    int kb   = (lane >> 4) * 8;
    int row  = min(row0 + r, n - 1);
    int nc   = w * 16 + r;

    v8f acc;
    float bv = bias[nc];
#pragma unroll
    for (int v = 0; v < 8; ++v) acc[v] = bv;

#pragma unroll
    for (int c = 0; c < (KPAD >> 5); ++c) {
        const int kc = c * 32;
        v16h a;
        const float* ap = A + (size_t)row * lda + kc + kb;

        // half 0: k in [kc+kb, kc+kb+8)   (compile-time bound checks)
        if (kc + 16 <= K) {
            float4 f0 = ((const float4*)ap)[0];
            float4 f1 = ((const float4*)ap)[1];
            a[0] = (_Float16)f0.x; a[1] = (_Float16)f0.y; a[2] = (_Float16)f0.z; a[3] = (_Float16)f0.w;
            a[4] = (_Float16)f1.x; a[5] = (_Float16)f1.y; a[6] = (_Float16)f1.z; a[7] = (_Float16)f1.w;
        } else {
#pragma unroll
            for (int j = 0; j < 8; ++j) a[j] = (_Float16)0.f;
        }
        // half 1: k in [kc+kb+16, kc+kb+24)
        if (kc + 32 <= K) {
            float4 f2 = ((const float4*)(ap + 16))[0];
            float4 f3 = ((const float4*)(ap + 16))[1];
            a[8]  = (_Float16)f2.x; a[9]  = (_Float16)f2.y; a[10] = (_Float16)f2.z; a[11] = (_Float16)f2.w;
            a[12] = (_Float16)f3.x; a[13] = (_Float16)f3.y; a[14] = (_Float16)f3.z; a[15] = (_Float16)f3.w;
        } else {
#pragma unroll
            for (int j = 0; j < 8; ++j) a[8 + j] = (_Float16)0.f;
        }

        const _Float16* bp = Bcm + (size_t)nc * KPAD + kc + kb;
        v8h b0 = *(const v8h*)bp;
        v8h b1 = *(const v8h*)(bp + 16);
        v16h b;
#pragma unroll
        for (int j = 0; j < 8; ++j) { b[j] = b0[j]; b[8 + j] = b1[j]; }

        acc = __builtin_amdgcn_wmma_f32_16x16x32_f16(false, a, false, b, (short)0, acc, false, false);
    }
    if (RELU) {
#pragma unroll
        for (int v = 0; v < 8; ++v) acc[v] = fmaxf(acc[v], 0.f);
    }
    int rbase = row0 + (lane >> 4) * 8;
#pragma unroll
    for (int v = 0; v < 8; ++v) {
        int rr = rbase + v;
        if (rr < n) C[(size_t)rr * ldc + nc] = acc[v];
    }
}

// ---------------------------------------------------------------------------
// Fused BiLSTM: one wave (32 threads) per 16 nodes.
// ---------------------------------------------------------------------------
__device__ __forceinline__ _Float16 l0w(const float* wih, const float* whh, int k, int nc) {
    if (k < 8)  return (_Float16)wih[nc * 8 + k];
    if (k < 24) return (_Float16)whh[nc * 16 + (k - 8)];
    return (_Float16)0.f;
}
__device__ __forceinline__ _Float16 l1w(const float* wih, const float* whh, int k, int nc) {
    if (k < 32) return (_Float16)wih[nc * 32 + k];
    if (k < 48) return (_Float16)whh[nc * 16 + (k - 32)];
    return (_Float16)0.f;
}

__global__ __launch_bounds__(32) void lstm_kernel(
    const float* __restrict__ temporal,
    const float* w0f_ih, const float* w0f_hh, const float* b0f,
    const float* w0b_ih, const float* w0b_hh, const float* b0b,
    const float* w1f_ih, const float* w1f_hh, const float* b1f,
    const float* w1b_ih, const float* w1b_hh, const float* b1b,
    float* __restrict__ g, int n) {
    __shared__ _Float16 l0out[TT][16][32];  // [step][node][fwd16|bwd16], rows 64B
    __shared__ _Float16 l1h[16][16];        // rows 32B

    int lane  = threadIdx.x;
    int node0 = blockIdx.x * 16;
    int r     = lane & 15;
    int kb    = (lane >> 4) * 8;
    int node  = min(node0 + r, n - 1);
    int rbase = (lane >> 4) * 8;

    // ---------------- layer 0 (both directions) ----------------
#pragma unroll 1
    for (int d = 0; d < 2; ++d) {
        const float* wih = d ? w0b_ih : w0f_ih;
        const float* whh = d ? w0b_hh : w0f_hh;
        const float* bb  = d ? b0b    : b0f;
        v16h bw[4];
        float biasv[4];
#pragma unroll
        for (int t = 0; t < 4; ++t) {
            int ncc = t * 16 + r;
            biasv[t] = bb[ncc];
#pragma unroll
            for (int j = 0; j < 8; ++j) {
                bw[t][j]     = l0w(wih, whh, kb + j,      ncc);
                bw[t][8 + j] = l0w(wih, whh, kb + 16 + j, ncc);
            }
        }
        v8f cell;
#pragma unroll
        for (int v = 0; v < 8; ++v) cell[v] = 0.f;

        bool rev = (d == 1);
#pragma unroll 1
        for (int si = 0; si < TT; ++si) {
            int s  = rev ? (TT - 1 - si) : si;
            int sp = rev ? min(s + 1, TT - 1) : max(s - 1, 0);
            v16h a;
            if (lane < 16) {
                const float4* xp = (const float4*)(temporal + ((size_t)node * TT + s) * 8);
                float4 x0 = xp[0], x1 = xp[1];
                a[0] = (_Float16)x0.x; a[1] = (_Float16)x0.y; a[2] = (_Float16)x0.z; a[3] = (_Float16)x0.w;
                a[4] = (_Float16)x1.x; a[5] = (_Float16)x1.y; a[6] = (_Float16)x1.z; a[7] = (_Float16)x1.w;
                if (si == 0) {
#pragma unroll
                    for (int j = 0; j < 8; ++j) a[8 + j] = (_Float16)0.f;
                } else {
                    v8h h = *(const v8h*)&l0out[sp][r][d * 16 + 8];
#pragma unroll
                    for (int j = 0; j < 8; ++j) a[8 + j] = h[j];
                }
            } else {
                if (si == 0) {
#pragma unroll
                    for (int j = 0; j < 8; ++j) a[j] = (_Float16)0.f;
                } else {
                    v8h h = *(const v8h*)&l0out[sp][r][d * 16];
#pragma unroll
                    for (int j = 0; j < 8; ++j) a[j] = h[j];
                }
#pragma unroll
                for (int j = 0; j < 8; ++j) a[8 + j] = (_Float16)0.f;
            }
            v8f gf[4];
#pragma unroll
            for (int t = 0; t < 4; ++t) {
                v8f ci;
#pragma unroll
                for (int v = 0; v < 8; ++v) ci[v] = biasv[t];
                gf[t] = __builtin_amdgcn_wmma_f32_16x16x32_f16(false, a, false, bw[t], (short)0, ci, false, false);
            }
#pragma unroll
            for (int v = 0; v < 8; ++v) {
                float ig = sigm(gf[0][v]);
                float fg = sigm(gf[1][v]);
                float gg = fast_tanh(gf[2][v]);
                float og = sigm(gf[3][v]);
                cell[v] = fg * cell[v] + ig * gg;
                float hv = og * fast_tanh(cell[v]);
                l0out[s][rbase + v][d * 16 + r] = (_Float16)hv;
            }
        }
    }

    // ---------------- layer 1 (both directions, only steps reaching t=5) ----
#pragma unroll 1
    for (int d = 0; d < 2; ++d) {
        const float* wih = d ? w1b_ih : w1f_ih;
        const float* whh = d ? w1b_hh : w1f_hh;
        const float* bb  = d ? b1b    : b1f;
        v16h bw[4][2];
        float biasv[4];
#pragma unroll
        for (int t = 0; t < 4; ++t) {
            int ncc = t * 16 + r;
            biasv[t] = bb[ncc];
#pragma unroll
            for (int c = 0; c < 2; ++c)
#pragma unroll
                for (int j = 0; j < 8; ++j) {
                    bw[t][c][j]     = l1w(wih, whh, c * 32 + kb + j,      ncc);
                    bw[t][c][8 + j] = l1w(wih, whh, c * 32 + kb + 16 + j, ncc);
                }
        }
        v8f cell;
#pragma unroll
        for (int v = 0; v < 8; ++v) cell[v] = 0.f;

        bool rev  = (d == 1);
        int steps = rev ? 19 : 6;  // fwd 0..5, bwd 23..5
#pragma unroll 1
        for (int si = 0; si < steps; ++si) {
            int s = rev ? (TT - 1 - si) : si;
            v16h a0, a1;
            const _Float16* fp = &l0out[s][r][0];
            v8h f0 = *(const v8h*)(fp + kb);
            v8h f1 = *(const v8h*)(fp + kb + 16);
#pragma unroll
            for (int j = 0; j < 8; ++j) { a0[j] = f0[j]; a0[8 + j] = f1[j]; }
            if (si == 0) {
#pragma unroll
                for (int j = 0; j < 8; ++j) a1[j] = (_Float16)0.f;
            } else {
                v8h h = *(const v8h*)&l1h[r][kb];
#pragma unroll
                for (int j = 0; j < 8; ++j) a1[j] = h[j];
            }
#pragma unroll
            for (int j = 0; j < 8; ++j) a1[8 + j] = (_Float16)0.f;

            v8f gf[4];
#pragma unroll
            for (int t = 0; t < 4; ++t) {
                v8f ci;
#pragma unroll
                for (int v = 0; v < 8; ++v) ci[v] = biasv[t];
                v8f c0 = __builtin_amdgcn_wmma_f32_16x16x32_f16(false, a0, false, bw[t][0], (short)0, ci, false, false);
                gf[t]  = __builtin_amdgcn_wmma_f32_16x16x32_f16(false, a1, false, bw[t][1], (short)0, c0, false, false);
            }
#pragma unroll
            for (int v = 0; v < 8; ++v) {
                float ig = sigm(gf[0][v]);
                float fg = sigm(gf[1][v]);
                float gg = fast_tanh(gf[2][v]);
                float og = sigm(gf[3][v]);
                cell[v] = fg * cell[v] + ig * gg;
                float hv = og * fast_tanh(cell[v]);
                l1h[rbase + v][r] = (_Float16)hv;
                if (s == 5) {
                    int nd = node0 + rbase + v;
                    if (nd < n) g[(size_t)nd * 160 + 128 + d * 16 + r] = hv;
                }
            }
        }
    }
}

// ---------------------------------------------------------------------------
// Edge / graph kernels
// ---------------------------------------------------------------------------
__global__ void edge_e_kernel(const float* __restrict__ ea, const float* __restrict__ ew,
                              const float* __restrict__ eb, float* __restrict__ e, int E) {
    int i = blockIdx.x * 256 + threadIdx.x;
    if (i >= E) return;
    const float4* ap = (const float4*)(ea + (size_t)i * 8);
    float4 a0 = ap[0], a1 = ap[1];
    const float4* wp = (const float4*)ew;
    float4 w0 = wp[0], w1 = wp[1];
    float s = eb[0] + a0.x * w0.x + a0.y * w0.y + a0.z * w0.z + a0.w * w0.w
                    + a1.x * w1.x + a1.y * w1.y + a1.z * w1.z + a1.w * w1.w;
    e[i] = __expf(s);
}

__global__ void zero_kernel(float* p, int cnt) {
    int i = blockIdx.x * 256 + threadIdx.x;
    if (i < cnt) p[i] = 0.f;
}

__global__ void denom_kernel(const int* __restrict__ src, const float* __restrict__ e,
                             float* __restrict__ denom, int E) {
    int i = blockIdx.x * 256 + threadIdx.x;
    if (i >= E) return;
    atomicAdd(&denom[src[i]], e[i]);
}

__global__ __launch_bounds__(256) void scatter_kernel(
    const int* __restrict__ src, const int* __restrict__ dst,
    const float* __restrict__ e, const float* __restrict__ denom,
    const float* __restrict__ x, float* __restrict__ acc, int E) {
    int lane = threadIdx.x & 31;
    int eidx = blockIdx.x * 8 + (threadIdx.x >> 5);
    if (eidx >= E) return;
    int s = src[eidx], d = dst[eidx];
    float w = e[eidx] / denom[s];
    const float4* xs = (const float4*)(x + (size_t)s * 128);
    float4 xv = xs[lane];
    float* ap = acc + (size_t)d * 128 + lane * 4;
    atomicAdd(ap + 0, w * xv.x);
    atomicAdd(ap + 1, w * xv.y);
    atomicAdd(ap + 2, w * xv.z);
    atomicAdd(ap + 3, w * xv.w);
}

__global__ void combine_kernel(const float* __restrict__ acc, const float* __restrict__ x,
                               float* __restrict__ out, int cnt) {
    int i = blockIdx.x * 256 + threadIdx.x;
    if (i < cnt) out[i] = fmaxf(acc[i] + x[i], 0.f);
}

__global__ void final_kernel(const float* __restrict__ g2, const float* __restrict__ ow,
                             const float* __restrict__ ob, float* __restrict__ out, int n) {
    int i = blockIdx.x * 256 + threadIdx.x;
    if (i >= n) return;
    float s = ob[0];
    const float4* gp = (const float4*)(g2 + (size_t)i * 128);
    const float4* wp = (const float4*)ow;
#pragma unroll
    for (int j = 0; j < 32; ++j) {
        float4 a = gp[j], b = wp[j];
        s += a.x * b.x + a.y * b.y + a.z * b.z + a.w * b.w;
    }
    out[i] = s;
}

// ---------------------------------------------------------------------------
extern "C" void kernel_launch(void* const* d_in, const int* in_sizes, int n_in,
                              void* d_out, int out_size, void* d_ws, size_t ws_size,
                              hipStream_t stream) {
    const float* x_sample = (const float*)d_in[0];
    const float* temporal = (const float*)d_in[1];
    const float* edge_attr= (const float*)d_in[2];
    const float* mlp_w1   = (const float*)d_in[3];
    const float* mlp_b1   = (const float*)d_in[4];
    const float* mlp_w2   = (const float*)d_in[5];
    const float* mlp_b2   = (const float*)d_in[6];
    const float* l0f_wih  = (const float*)d_in[7];
    const float* l0f_whh  = (const float*)d_in[8];
    const float* l0f_b    = (const float*)d_in[9];
    const float* l0b_wih  = (const float*)d_in[10];
    const float* l0b_whh  = (const float*)d_in[11];
    const float* l0b_b    = (const float*)d_in[12];
    const float* l1f_wih  = (const float*)d_in[13];
    const float* l1f_whh  = (const float*)d_in[14];
    const float* l1f_b    = (const float*)d_in[15];
    const float* l1b_wih  = (const float*)d_in[16];
    const float* l1b_whh  = (const float*)d_in[17];
    const float* l1b_b    = (const float*)d_in[18];
    const float* g0_alpha = (const float*)d_in[19];
    const float* g0_beta  = (const float*)d_in[20];
    const float* g0_featW = (const float*)d_in[21];
    const float* g0_edgeW = (const float*)d_in[22];
    const float* g0_edgeB = (const float*)d_in[23];
    const float* g1_alpha = (const float*)d_in[24];
    const float* g1_beta  = (const float*)d_in[25];
    const float* g1_featW = (const float*)d_in[26];
    const float* g1_edgeW = (const float*)d_in[27];
    const float* g1_edgeB = (const float*)d_in[28];
    const float* out_w    = (const float*)d_in[29];
    const float* out_b    = (const float*)d_in[30];
    const int*   eidx     = (const int*)d_in[31];
    const int*   src      = eidx;
    const int*   dst      = eidx + EE;

    char* ws = (char*)d_ws;
    float*     gbuf  = (float*)(ws + 0);            // N x 160
    float*     t1    = (float*)(ws + 32000000);     // N x 128 (mlp hidden / conv acc)
    float*     t2    = (float*)(ws + 57600000);     // N x 128 (conv transform x)
    float*     t3    = (float*)(ws + 83200000);     // N x 128 (conv0 output)
    float*     ebuf  = (float*)(ws + 108800000);    // E
    float*     denom = (float*)(ws + 111200000);    // N
    _Float16*  wb1   = (_Float16*)(ws + 111400000); // 128 x 32
    _Float16*  wb2   = (_Float16*)(ws + 111408192); // 128 x 128
    _Float16*  wA0   = (_Float16*)(ws + 111440960); // 128 x 160
    _Float16*  wA1   = (_Float16*)(ws + 111481920); // 128 x 128

    const int N = NN, E = EE;
    const int rowBlocks = (N + 15) / 16;            // 3125
    const int eBlocks   = (E + 255) / 256;

    // weight prep
    pack_wkn_kernel<<<(32 * 128 + 255) / 256, 256, 0, stream>>>(mlp_w1, wb1, 16, 32);
    pack_wkn_kernel<<<(128 * 128 + 255) / 256, 256, 0, stream>>>(mlp_w2, wb2, 128, 128);
    alphaW_pack_kernel<<<(160 * 128 + 255) / 256, 256, 0, stream>>>(g0_alpha, g0_featW, wA0, 160, 160);
    alphaW_pack_kernel<<<(128 * 128 + 255) / 256, 256, 0, stream>>>(g1_alpha, g1_featW, wA1, 128, 128);

    // MLP -> g[:, 0:128]
    gemm_kernel<16, 32, true><<<rowBlocks, 256, 0, stream>>>(x_sample, 16, wb1, mlp_b1, t1, 128, N);
    gemm_kernel<128, 128, false><<<rowBlocks, 256, 0, stream>>>(t1, 128, wb2, mlp_b2, gbuf, 160, N);

    // BiLSTM -> g[:, 128:160]
    lstm_kernel<<<rowBlocks, 32, 0, stream>>>(temporal,
        l0f_wih, l0f_whh, l0f_b, l0b_wih, l0b_whh, l0b_b,
        l1f_wih, l1f_whh, l1f_b, l1b_wih, l1b_whh, l1b_b, gbuf, N);

    // conv0: x = g @ (alpha@featW) + beta
    gemm_kernel<160, 160, false><<<rowBlocks, 256, 0, stream>>>(gbuf, 160, wA0, g0_beta, t2, 128, N);
    edge_e_kernel<<<eBlocks, 256, 0, stream>>>(edge_attr, g0_edgeW, g0_edgeB, ebuf, E);
    zero_kernel<<<(N + 255) / 256, 256, 0, stream>>>(denom, N);
    denom_kernel<<<eBlocks, 256, 0, stream>>>(src, ebuf, denom, E);
    zero_kernel<<<(N * 128 + 255) / 256, 256, 0, stream>>>(t1, N * 128);
    scatter_kernel<<<(E + 7) / 8, 256, 0, stream>>>(src, dst, ebuf, denom, t2, t1, E);
    combine_kernel<<<(N * 128 + 255) / 256, 256, 0, stream>>>(t1, t2, t3, N * 128);

    // conv1
    gemm_kernel<128, 128, false><<<rowBlocks, 256, 0, stream>>>(t3, 128, wA1, g1_beta, t2, 128, N);
    edge_e_kernel<<<eBlocks, 256, 0, stream>>>(edge_attr, g1_edgeW, g1_edgeB, ebuf, E);
    zero_kernel<<<(N + 255) / 256, 256, 0, stream>>>(denom, N);
    denom_kernel<<<eBlocks, 256, 0, stream>>>(src, ebuf, denom, E);
    zero_kernel<<<(N * 128 + 255) / 256, 256, 0, stream>>>(t1, N * 128);
    scatter_kernel<<<(E + 7) / 8, 256, 0, stream>>>(src, dst, ebuf, denom, t2, t1, E);
    combine_kernel<<<(N * 128 + 255) / 256, 256, 0, stream>>>(t1, t2, gbuf, N * 128); // g2, stride 128

    // output head
    final_kernel<<<(N + 255) / 256, 256, 0, stream>>>(gbuf, out_w, out_b, (float*)d_out, N);
}
